// HGNNLayer_28939489641060
// MI455X (gfx1250) — matrix operand order repaired and checked by reading, not compiled, not run on hardware
//
#include <hip/hip_runtime.h>
#include <hip/hip_bf16.h>

typedef __bf16 v16bf __attribute__((ext_vector_type(16)));
typedef float  v8f   __attribute__((ext_vector_type(8)));

#define B_ 4
#define N_ 4096
#define E_ 4096
#define D_ 128

__device__ __forceinline__ unsigned short f2bf(float f) {
    unsigned u = __float_as_uint(f);
    u += 0x7FFFu + ((u >> 16) & 1u);         // round-to-nearest-even
    return (unsigned short)(u >> 16);
}

__device__ __forceinline__ uint4 pack8(float4 a, float4 b) {
    uint4 q;
    q.x = f2bf(a.x) | ((unsigned)f2bf(a.y) << 16);
    q.y = f2bf(a.z) | ((unsigned)f2bf(a.w) << 16);
    q.z = f2bf(b.x) | ((unsigned)f2bf(b.y) << 16);
    q.w = f2bf(b.z) | ((unsigned)f2bf(b.w) << 16);
    return q;
}

enum { A_BF16 = 0, A_F32 = 1 };
enum { B_NT_F32 = 0, B_TR_F32 = 1, B_BF16ROW = 2 };
enum { EPI_F32 = 0, EPI_BF16 = 1, EPI_EPSADD = 2, EPI_LEAKYBN = 3 };

// C_tile[128 x 128] = A[128,K](bf16 or f32) x B[K,128]; fp32 accumulate via
// v_wmma_f32_16x16x32_bf16. 8 waves, each wave: 16 rows x 128 cols.
// Software-pipelined: global fetch of tile k+1 overlaps WMMA of tile k.
template <int AK, int BK, int EPI, int HTT, int MLONG>
__launch_bounds__(256)
__global__ void gemm_wmma(
    const unsigned short* __restrict__ Abf, const float* __restrict__ Af, int lda, long sA,
    const float* __restrict__ Bf, const unsigned short* __restrict__ Bh, int ldb, long sB,
    float* __restrict__ Cf, unsigned short* __restrict__ Cb, int ldc, long sC,
    int Ktot, int Nout,
    const float* __restrict__ xtheta, float* __restrict__ httheta,
    const float* __restrict__ htxwT, const float* __restrict__ epsp,
    const float* __restrict__ gamma, const float* __restrict__ beta,
    const float* __restrict__ mean,  const float* __restrict__ var)
{
    __shared__ unsigned short ldsA[128 * 40];  // pitch 40 halves = 80B (bank-friendly)
    __shared__ unsigned short ldsB[128 * 40];
    __shared__ float httl[128];

    const long bb = blockIdx.y;
    const int  m0 = MLONG ? blockIdx.x * 128 : 0;
    const int  n0 = MLONG ? 0 : blockIdx.x * 128;

    if (AK == A_BF16) Abf += bb * sA; else Af += bb * sA;
    if (BK == B_BF16ROW) Bh += bb * sB; else Bf += bb * sB;
    Cf += bb * sC; Cb += bb * sC;
    if (HTT) { xtheta += bb * (long)Ktot; httheta += bb * (long)Nout; }
    if (EPI == EPI_EPSADD) htxwT += bb * (long)128 * ldc;

    const int t    = threadIdx.x;
    const int w    = t >> 5;
    const int lane = t & 31;
    const int lrow = lane & 15;
    const int g2   = lane >> 4;

    // staging thread coordinates (fixed per thread)
    const int arow = t >> 1, akc = t & 1;            // A: row-copy
    const int kb = (t >> 5) * 4, cb4 = (t & 31) * 4; // B NT: 4x4 micro-transpose
    const int bcol = t >> 1, bkc = t & 1;            // B TR / BF16: row-copy

    v8f acc[8];
#pragma unroll
    for (int ct = 0; ct < 8; ++ct)
#pragma unroll
        for (int i = 0; i < 8; ++i) acc[ct][i] = 0.f;

    float tp[4] = {0.f, 0.f, 0.f, 0.f};
    if (HTT && t < 128) httl[t] = 0.f;

    // fetch registers (raw global data; converted at commit time)
    uint4  qa0, qa1;                 // A bf16 path
    float4 fa0, fa1, fa2, fa3;       // A f32 path
    float4 rb0, rb1, rb2, rb3;       // B f32 paths
    uint4  qb0, qb1;                 // B bf16 path
    float  xv0 = 0.f, xv1 = 0.f, xv2 = 0.f, xv3 = 0.f;

    auto FETCH = [&](int k0) {
        if (AK == A_BF16) {
            const uint4* g = reinterpret_cast<const uint4*>(Abf + (long)arow * lda + k0 + akc * 16);
            qa0 = g[0]; qa1 = g[1];
        } else {
            const float4* g = reinterpret_cast<const float4*>(Af + (long)(m0 + arow) * lda + k0 + akc * 16);
            fa0 = g[0]; fa1 = g[1]; fa2 = g[2]; fa3 = g[3];
        }
        if (BK == B_NT_F32) {
            rb0 = *reinterpret_cast<const float4*>(Bf + (long)(k0 + kb + 0) * ldb + n0 + cb4);
            rb1 = *reinterpret_cast<const float4*>(Bf + (long)(k0 + kb + 1) * ldb + n0 + cb4);
            rb2 = *reinterpret_cast<const float4*>(Bf + (long)(k0 + kb + 2) * ldb + n0 + cb4);
            rb3 = *reinterpret_cast<const float4*>(Bf + (long)(k0 + kb + 3) * ldb + n0 + cb4);
            if (HTT) {
                xv0 = xtheta[k0 + kb + 0]; xv1 = xtheta[k0 + kb + 1];
                xv2 = xtheta[k0 + kb + 2]; xv3 = xtheta[k0 + kb + 3];
            }
        } else if (BK == B_TR_F32) {
            const float4* g = reinterpret_cast<const float4*>(Bf + (long)(n0 + bcol) * ldb + k0 + bkc * 16);
            rb0 = g[0]; rb1 = g[1]; rb2 = g[2]; rb3 = g[3];
        } else {
            const uint4* g = reinterpret_cast<const uint4*>(Bh + (long)bcol * Ktot + k0 + bkc * 16);
            qb0 = g[0]; qb1 = g[1];
        }
    };

    auto COMMIT = [&]() {
        // A tile -> ldsA[row][k], row-major
        {
            uint4 q0, q1;
            if (AK == A_BF16) { q0 = qa0; q1 = qa1; }
            else { q0 = pack8(fa0, fa1); q1 = pack8(fa2, fa3); }
            uint4* s = reinterpret_cast<uint4*>(&ldsA[arow * 40 + akc * 16]);
            s[0] = q0; s[1] = q1;
        }
        // B tile -> ldsB[col][k], column-major within tile
        if (BK == B_NT_F32) {
            float a0[4] = {rb0.x, rb0.y, rb0.z, rb0.w};
            float a1[4] = {rb1.x, rb1.y, rb1.z, rb1.w};
            float a2[4] = {rb2.x, rb2.y, rb2.z, rb2.w};
            float a3[4] = {rb3.x, rb3.y, rb3.z, rb3.w};
#pragma unroll
            for (int j = 0; j < 4; ++j) {
                if (HTT) tp[j] += a0[j] * xv0 + a1[j] * xv1 + a2[j] * xv2 + a3[j] * xv3;
                uint2 p;
                p.x = f2bf(a0[j]) | ((unsigned)f2bf(a1[j]) << 16);
                p.y = f2bf(a2[j]) | ((unsigned)f2bf(a3[j]) << 16);
                *reinterpret_cast<uint2*>(&ldsB[(cb4 + j) * 40 + kb]) = p;
            }
        } else if (BK == B_TR_F32) {
            uint4* s = reinterpret_cast<uint4*>(&ldsB[bcol * 40 + bkc * 16]);
            s[0] = pack8(rb0, rb1); s[1] = pack8(rb2, rb3);
        } else {
            uint4* s = reinterpret_cast<uint4*>(&ldsB[bcol * 40 + bkc * 16]);
            s[0] = qb0; s[1] = qb1;
        }
    };

    const int nIter = Ktot >> 5;
    FETCH(0);

    const unsigned short* arowp = &ldsA[(w * 16 + lrow) * 40];

    for (int it = 0; it < nIter; ++it) {
        COMMIT();                    // waits on in-flight loads, converts, fills LDS
        __syncthreads();
        if (it + 1 < nIter) FETCH((it + 1) << 5);   // next tile's loads in flight

        union Frag { v16bf v; uint4 q[2]; };
        Frag af;
        af.q[0] = *reinterpret_cast<const uint4*>(arowp + 8 * g2);       // K h<8
        af.q[1] = *reinterpret_cast<const uint4*>(arowp + 16 + 8 * g2);  // K h>=8
#pragma unroll
        for (int ct = 0; ct < 8; ++ct) {
            Frag bf_;
            const unsigned short* brow = &ldsB[(ct * 16 + lrow) * 40 + 8 * g2];
            bf_.q[0] = *reinterpret_cast<const uint4*>(brow);
            bf_.q[1] = *reinterpret_cast<const uint4*>(brow + 16);
            acc[ct] = __builtin_amdgcn_wmma_f32_16x16x32_bf16(
                false, af.v, false, bf_.v, (short)0, acc[ct], false, false);
        }
        __syncthreads();
    }

    // ---- fused incident^T @ (x@theta) reduction (G2 only) ----
    if (HTT) {
#pragma unroll
        for (int j = 0; j < 4; ++j) atomicAdd(&httl[cb4 + j], tp[j]);
        __syncthreads();
        if (t < 128) httheta[n0 + t] = httl[t];
    }

    // ---- epilogue ----
    const float eps0 = (EPI == EPI_EPSADD) ? epsp[0] : 0.f;
#pragma unroll
    for (int ct = 0; ct < 8; ++ct) {
        const int col = n0 + ct * 16 + lrow;
        float gam = 0.f, bet = 0.f, mu = 0.f, rv = 0.f;
        if (EPI == EPI_LEAKYBN) {
            const int d = ct * 16 + lrow;
            gam = gamma[d]; bet = beta[d]; mu = mean[d];
            rv = rsqrtf(var[d] + 1e-5f);
        }
#pragma unroll
        for (int r = 0; r < 8; ++r) {
            const int mrow = w * 16 + r + 8 * g2;
            float v = acc[ct][r];
            if (EPI == EPI_F32) {
                Cf[(long)mrow * ldc + col] = v;
            } else if (EPI == EPI_BF16) {
                Cb[(long)mrow * ldc + col] = f2bf(v);
            } else if (EPI == EPI_EPSADD) {
                Cb[(long)mrow * ldc + col] = f2bf(v + eps0 * htxwT[(long)mrow * ldc + col]);
            } else {
                v = (v >= 0.f) ? v : 0.01f * v;                 // leaky relu
                v = (v - mu) * rv * gam + bet;                  // BN inference
                Cf[(long)(m0 + mrow) * 128 + (ct * 16 + lrow)] = v;
            }
        }
    }
}

// xwT[b][d][n] = (x@W + b)^T as bf16 ; xtheta[b][n] = x@theta
__global__ __launch_bounds__(128) void k_mlp(
    const float* __restrict__ x, const float* __restrict__ W,
    const float* __restrict__ bias, const float* __restrict__ theta,
    unsigned short* __restrict__ xwT, float* __restrict__ xtheta)
{
    __shared__ float xs[128];
    __shared__ float red[128];
    const int n = blockIdx.x, b = blockIdx.y, d = threadIdx.x;
    xs[d] = x[((long)b * N_ + n) * D_ + d];
    __syncthreads();
    float acc = bias[d];
#pragma unroll 8
    for (int k = 0; k < 128; ++k) acc += xs[k] * W[k * 128 + d];
    xwT[((long)b * 128 + d) * N_ + n] = f2bf(acc);
    red[d] = xs[d] * theta[d];
    __syncthreads();
    for (int s = 64; s > 0; s >>= 1) {
        if (d < s) red[d] += red[d + s];
        __syncthreads();
    }
    if (d == 0) xtheta[(long)b * N_ + n] = red[0];
}

__global__ __launch_bounds__(256) void k_softmax(
    const float* __restrict__ httheta, const int* __restrict__ emask,
    float* __restrict__ attn)
{
    __shared__ float red[256];
    const int b = blockIdx.x, t = threadIdx.x;
    const float* s = httheta + (long)b * E_;
    const int* m = emask + (long)b * E_;
    float mx = -3.4e38f;
    for (int e = t; e < E_; e += 256) if (m[e] != 0) mx = fmaxf(mx, s[e]);
    red[t] = mx; __syncthreads();
    for (int st = 128; st > 0; st >>= 1) { if (t < st) red[t] = fmaxf(red[t], red[t + st]); __syncthreads(); }
    mx = red[0]; __syncthreads();
    float sm = 0.f;
    for (int e = t; e < E_; e += 256) if (m[e] != 0) sm += __expf(s[e] - mx);
    red[t] = sm; __syncthreads();
    for (int st = 128; st > 0; st >>= 1) { if (t < st) red[t] += red[t + st]; __syncthreads(); }
    const float inv = 1.f / red[0];
    for (int e = t; e < E_; e += 256)
        attn[(long)b * E_ + e] = (m[e] != 0) ? __expf(s[e] - mx) * inv : 0.f;
}

// b3T[b][d][e] = bf16(attn[b][e] * htxwT[b][d][e])
__global__ __launch_bounds__(256) void k_scale(
    const float* __restrict__ htxwT, const float* __restrict__ attn,
    unsigned short* __restrict__ b3T)
{
    const long i = (long)blockIdx.x * 256 + threadIdx.x;  // B*128*E total
    const int col = (int)(i & (E_ - 1));
    const int b = (int)(i >> 19);                         // 128*4096 = 2^19
    b3T[i] = f2bf(attn[(long)b * E_ + col] * htxwT[i]);
}

extern "C" void kernel_launch(void* const* d_in, const int* in_sizes, int n_in,
                              void* d_out, int out_size, void* d_ws, size_t ws_size,
                              hipStream_t stream) {
    const float* inc   = (const float*)d_in[0];
    const float* degv  = (const float*)d_in[1];
    const float* dege  = (const float*)d_in[2];
    const float* x     = (const float*)d_in[3];
    const int*   emask = (const int*)d_in[4];
    const float* W     = (const float*)d_in[5];
    const float* bias  = (const float*)d_in[6];
    const float* theta = (const float*)d_in[7];
    const float* epsp  = (const float*)d_in[8];
    const float* gamma = (const float*)d_in[9];
    const float* beta  = (const float*)d_in[10];
    const float* mean  = (const float*)d_in[11];
    const float* var   = (const float*)d_in[12];
    float* out = (float*)d_out;

    char* w = (char*)d_ws;
    unsigned short* xwT   = (unsigned short*)w; w += (size_t)B_ * 128 * N_ * 2;
    float* xtheta         = (float*)w;          w += (size_t)B_ * N_ * 4;
    float* htxwT          = (float*)w;          w += (size_t)B_ * 128 * E_ * 4;
    float* httheta        = (float*)w;          w += (size_t)B_ * E_ * 4;
    float* attn           = (float*)w;          w += (size_t)B_ * E_ * 4;
    unsigned short* b3T   = (unsigned short*)w; w += (size_t)B_ * 128 * E_ * 2;
    unsigned short* tA    = (unsigned short*)w; w += (size_t)B_ * 128 * N_ * 2;
    unsigned short* tB    = (unsigned short*)w; w += (size_t)B_ * 128 * E_ * 2;

    const dim3 blk(256);
    const dim3 g32(32, B_);

    k_mlp<<<dim3(N_, B_), 128, 0, stream>>>(x, W, bias, theta, xwT, xtheta);

    // G2: htxwT(f32) = xwT @ inc ; fused httheta = inc^T(x@theta)
    gemm_wmma<A_BF16, B_NT_F32, EPI_F32, 1, 0><<<g32, blk, 0, stream>>>(
        xwT, nullptr, N_, (long)128 * N_,
        inc, nullptr, E_, (long)N_ * E_,
        htxwT, nullptr, E_, (long)128 * E_,
        N_, E_, xtheta, httheta, nullptr, nullptr,
        nullptr, nullptr, nullptr, nullptr);

    k_softmax<<<B_, 256, 0, stream>>>(httheta, emask, attn);
    k_scale<<<(B_ * 128 * E_) / 256, 256, 0, stream>>>(htxwT, attn, b3T);

    // G3: h1T = b3T @ inc^T
    gemm_wmma<A_BF16, B_TR_F32, EPI_BF16, 0, 0><<<g32, blk, 0, stream>>>(
        b3T, nullptr, E_, (long)128 * E_,
        inc, nullptr, E_, (long)N_ * E_,
        nullptr, tA, N_, (long)128 * N_,
        E_, N_, nullptr, nullptr, nullptr, nullptr,
        nullptr, nullptr, nullptr, nullptr);

    // G4: h2T = h1T @ degv^T
    gemm_wmma<A_BF16, B_TR_F32, EPI_BF16, 0, 0><<<g32, blk, 0, stream>>>(
        tA, nullptr, N_, (long)128 * N_,
        degv, nullptr, N_, (long)N_ * N_,
        nullptr, tB, N_, (long)128 * N_,
        N_, N_, nullptr, nullptr, nullptr, nullptr,
        nullptr, nullptr, nullptr, nullptr);

    // G5: h3T = h2T @ inc
    gemm_wmma<A_BF16, B_NT_F32, EPI_BF16, 0, 0><<<g32, blk, 0, stream>>>(
        tB, nullptr, N_, (long)128 * N_,
        inc, nullptr, E_, (long)N_ * E_,
        nullptr, tA, E_, (long)128 * E_,
        N_, E_, nullptr, nullptr, nullptr, nullptr,
        nullptr, nullptr, nullptr, nullptr);

    // G6: hT = bf16(h3T @ dege^T + eps * htxwT)
    gemm_wmma<A_BF16, B_TR_F32, EPI_EPSADD, 0, 0><<<g32, blk, 0, stream>>>(
        tA, nullptr, E_, (long)128 * E_,
        dege, nullptr, E_, (long)E_ * E_,
        nullptr, tB, E_, (long)128 * E_,
        E_, E_, nullptr, nullptr, htxwT, epsp,
        nullptr, nullptr, nullptr, nullptr);

    // G7: out = BN(leaky(inc @ h)) in natural [B,N,D] layout
    gemm_wmma<A_F32, B_BF16ROW, EPI_LEAKYBN, 0, 1><<<g32, blk, 0, stream>>>(
        nullptr, inc, E_, (long)N_ * E_,
        nullptr, tB, E_, (long)128 * E_,
        out, nullptr, 128, (long)N_ * 128,
        E_, 128, nullptr, nullptr, nullptr, nullptr,
        gamma, beta, mean, var);
}
